// MoELayer_81561428951090
// MI455X (gfx1250) — matrix operand order repaired and checked by reading, not compiled, not run on hardware
//
#include <hip/hip_runtime.h>
#include <hip/hip_bf16.h>
#include <math.h>

// Problem constants (from reference): H=512, FF=4H=2048, E=8, K=2, B*S=16384
#define H     512
#define FF    2048
#define NE    8
#define NTOK  16384
#define MT    64      // tokens per expert tile
#define FC    128     // FF chunk width
#define NCH   (FF / FC)   // 16 chunks
#define XS_S  520     // x tile LDS row stride (bf16 elems), padded for banks
#define HB_S  136     // h buffer LDS row stride
#define WP_S  34      // weight panel LDS col stride (32 k + 2 pad)

typedef __attribute__((ext_vector_type(16))) __bf16 v16bf;
typedef __attribute__((ext_vector_type(8)))  float  v8f;

union ABFrag { v16bf v; unsigned int u[8]; };
union CDFrag { v8f   v; float        f[8]; };

__device__ __forceinline__ unsigned short f2bf(float f) {
    unsigned int u = __builtin_bit_cast(unsigned int, f);
    unsigned int r = u + 0x7fffu + ((u >> 16) & 1u);   // round-to-nearest-even
    return (unsigned short)(r >> 16);
}

// A fragment: 16x32 bf16, row-major LDS (stride in ushorts folded into rowBase)
__device__ __forceinline__ void load_a(ABFrag& a, const unsigned short* lds,
                                       int rowBase, int k0, int half) {
#pragma unroll
    for (int v = 0; v < 8; ++v) {
        int ko = ((v < 4) ? 2 * v : 16 + 2 * (v - 4)) + 8 * half;
        a.u[v] = *(const unsigned int*)(lds + rowBase + k0 + ko);
    }
}

// B fragment: 32x16 bf16, col-major LDS panel [n][k], stride WP_S
__device__ __forceinline__ void load_b(ABFrag& b, const unsigned short* lds,
                                       int colBase, int half) {
#pragma unroll
    for (int v = 0; v < 8; ++v) {
        int ko = ((v < 4) ? 2 * v : 16 + 2 * (v - 4)) + 8 * half;
        b.u[v] = *(const unsigned int*)(lds + colBase + ko);
    }
}

__device__ __forceinline__ void wmma_bf16(CDFrag& c, const ABFrag& a, const ABFrag& b) {
    c.v = __builtin_amdgcn_wmma_f32_16x16x32_bf16(false, a.v, false, b.v,
                                                  (short)0, c.v, false, false);
}

__device__ __forceinline__ float fast_silu(float v) {
    // v * sigmoid(v) with HW rcp (1 ulp-ish; inputs are bf16-grade anyway)
    return v * __builtin_amdgcn_rcpf(1.f + __expf(-v));
}

// ---------------------------------------------------------------------------
__global__ void k_zero(int* p, int n) {
    int i = threadIdx.x;
    if (i < n) p[i] = 0;
}

__global__ void k_f32_to_bf16(const float* __restrict__ src,
                              unsigned short* __restrict__ dst, int n4) {
    int i = blockIdx.x * blockDim.x + threadIdx.x;
    if (i >= n4) return;
    float4 d = ((const float4*)src)[i];
    unsigned int lo = (unsigned int)f2bf(d.x) | ((unsigned int)f2bf(d.y) << 16);
    unsigned int hi = (unsigned int)f2bf(d.z) | ((unsigned int)f2bf(d.w) << 16);
    uint2 o; o.x = lo; o.y = hi;
    ((uint2*)dst)[i] = o;
}

// ---------------------------------------------------------------------------
// Gate: one wave per token. Computes top-2 routing, scatters (token, weight)
// into per-expert lists, converts x row to bf16, seeds out with weighted b2.
__global__ __launch_bounds__(256)
void k_gate(const float* __restrict__ x,  const float* __restrict__ Wg,
            const float* __restrict__ bg, const float* __restrict__ b2,
            unsigned short* __restrict__ xbf,
            int* __restrict__ tok_list, float* __restrict__ w_list,
            int* __restrict__ cnt, float* __restrict__ out)
{
    const int tid  = threadIdx.x;
    const int wave = tid >> 5;
    const int lane = tid & 31;
    const int tok  = blockIdx.x * 8 + wave;

    float acc[NE];
#pragma unroll
    for (int e = 0; e < NE; ++e) acc[e] = 0.f;

    const float* xr = x + (size_t)tok * H;
#pragma unroll
    for (int i = 0; i < H / 32; ++i) {
        int h = lane + 32 * i;
        float xv = xr[h];
        xbf[(size_t)tok * H + h] = f2bf(xv);
        const float* wr = Wg + (size_t)h * NE;
#pragma unroll
        for (int e = 0; e < NE; ++e) acc[e] += xv * wr[e];
    }
#pragma unroll
    for (int off = 16; off >= 1; off >>= 1)
#pragma unroll
        for (int e = 0; e < NE; ++e) acc[e] += __shfl_xor(acc[e], off, 32);

    float lg[NE];
#pragma unroll
    for (int e = 0; e < NE; ++e) lg[e] = acc[e] + bg[e];

    int e0 = 0; float l0 = lg[0];
#pragma unroll
    for (int e = 1; e < NE; ++e) if (lg[e] > l0) { l0 = lg[e]; e0 = e; }
    int e1 = (e0 == 0) ? 1 : 0; float l1 = lg[e1];
#pragma unroll
    for (int e = 0; e < NE; ++e) if (e != e0 && lg[e] > l1) { l1 = lg[e]; e1 = e; }

    float t  = __expf(l1 - l0);           // softmax over the two top logits
    float w0 = __builtin_amdgcn_rcpf(1.f + t);
    float w1 = t * w0;

    if (lane == 0) {
        int s0 = atomicAdd(&cnt[e0], 1);
        tok_list[e0 * NTOK + s0] = tok;
        w_list [e0 * NTOK + s0] = w0;
        int s1 = atomicAdd(&cnt[e1], 1);
        tok_list[e1 * NTOK + s1] = tok;
        w_list [e1 * NTOK + s1] = w1;
    }

    const float* b2a = b2 + (size_t)e0 * H;
    const float* b2b = b2 + (size_t)e1 * H;
    float* orow = out + (size_t)tok * H;
#pragma unroll
    for (int i = 0; i < H / 32; ++i) {
        int h = lane + 32 * i;
        orow[h] = w0 * b2a[h] + w1 * b2b[h];
    }
}

// ---------------------------------------------------------------------------
// Fused expert FFN over a 64-token gathered tile.
// GEMM1: h = silu(x @ W1[:, chunk] + b1) * route_w   (bf16 -> LDS)
// GEMM2: out_acc += h @ W2[chunk, :]                 (f32 VGPR accumulators)
__global__ __launch_bounds__(256, 1)
void k_expert(const unsigned short* __restrict__ xbf,
              const unsigned short* __restrict__ w1bf,
              const unsigned short* __restrict__ w2bf,
              const float* __restrict__ b1,
              const int*   __restrict__ tok_list,
              const float* __restrict__ w_list,
              const int*   __restrict__ cnt,
              float* __restrict__ out, int e)
{
    __shared__ __align__(16) unsigned short xs[MT * XS_S];      // 66,560 B
    __shared__ __align__(16) unsigned short hb[MT * HB_S];      // 17,408 B
    __shared__ __align__(16) unsigned short w1p[2][FC * WP_S];  // 17,408 B
    __shared__ __align__(16) unsigned short w2p[H * WP_S];      // 34,816 B
    __shared__ __align__(16) float wrow[MT];
    __shared__ __align__(16) int   trow[MT];

    const int tid  = threadIdx.x;
    const int wave = tid >> 5;
    const int lane = tid & 31;
    const int half = lane >> 4;
    const int ln16 = lane & 15;

    const int nInList = cnt[e];
    const int base = blockIdx.x * MT;
    if (base >= nInList) return;          // uniform exit: empty tile

    // ---- per-wave b1 column values for all 16 chunks (registers) ----
    float b1reg[NCH];
#pragma unroll
    for (int c = 0; c < NCH; ++c)
        b1reg[c] = b1[(size_t)e * FF + c * FC + wave * 16 + ln16];

    // ---- gather x tile (bf16 rows) + per-row routing weight ----
    {
        int r = tid >> 2, seg = tid & 3;
        int li  = base + r;
        int tok = (li < nInList) ? tok_list[e * NTOK + li] : -1;
        if (seg == 0) {
            trow[r] = tok;
            wrow[r] = (tok >= 0) ? w_list[e * NTOK + li] : 0.f;
        }
        uint4* dst = (uint4*)(xs + r * XS_S + seg * 128);
        if (tok >= 0) {
            const uint4* src = (const uint4*)(xbf + (size_t)tok * H + seg * 128);
#pragma unroll
            for (int i = 0; i < 16; ++i) dst[i] = src[i];
        } else {
            uint4 z; z.x = z.y = z.z = z.w = 0u;
#pragma unroll
            for (int i = 0; i < 16; ++i) dst[i] = z;
        }
    }

    CDFrag acc2[4][4];                    // wave out tile: 64 rows x 64 cols
#pragma unroll
    for (int mt = 0; mt < 4; ++mt)
#pragma unroll
        for (int nt = 0; nt < 4; ++nt)
#pragma unroll
            for (int r = 0; r < 8; ++r) acc2[mt][nt].f[r] = 0.f;

    const int w1nblk = tid & 31, w1krow = tid >> 5;       // W1 stage mapping
    const int w2nblk = tid & 127, w2kr  = tid >> 7;       // W2 stage mapping

    for (int c = 0; c < NCH; ++c) {
        const int ff0 = c * FC;

        // ---------------- GEMM1 over K = H (double-buffered W1 panel) ------
        CDFrag acc1[4];
#pragma unroll
        for (int mt = 0; mt < 4; ++mt)
#pragma unroll
            for (int r = 0; r < 8; ++r) acc1[mt].f[r] = 0.f;

        // stage k-step 0 into buffer 0 (safe: all prior readers barriered)
        {
            uint2 d[4];
#pragma unroll
            for (int i = 0; i < 4; ++i) {
                const unsigned short* src =
                    w1bf + ((size_t)e * H + (size_t)(w1krow + 8 * i)) * FF
                         + ff0 + w1nblk * 4;
                d[i] = *(const uint2*)src;
                __builtin_prefetch(src + 32 * FF, 0, 3);   // next k-panel, WGP$
            }
#pragma unroll
            for (int i = 0; i < 4; ++i) {
                int k = w1krow + 8 * i;
                const unsigned short* ds = (const unsigned short*)&d[i];
#pragma unroll
                for (int j = 0; j < 4; ++j)
                    w1p[0][(w1nblk * 4 + j) * WP_S + k] = ds[j];
            }
        }

        for (int kk = 0; kk < H / 32; ++kk) {
            __syncthreads();   // staged kk visible; buf (kk+1)&1 free to write
            if (kk + 1 < H / 32) {
                uint2 d[4];
#pragma unroll
                for (int i = 0; i < 4; ++i) {
                    const unsigned short* src =
                        w1bf + ((size_t)e * H + (size_t)((kk + 1) * 32 + w1krow + 8 * i)) * FF
                             + ff0 + w1nblk * 4;
                    d[i] = *(const uint2*)src;
                    __builtin_prefetch(src + 32 * FF, 0, 3);
                }
#pragma unroll
                for (int i = 0; i < 4; ++i) {
                    int k = w1krow + 8 * i;
                    const unsigned short* ds = (const unsigned short*)&d[i];
#pragma unroll
                    for (int j = 0; j < 4; ++j)
                        w1p[(kk + 1) & 1][(w1nblk * 4 + j) * WP_S + k] = ds[j];
                }
            }
            const unsigned short* wcur = w1p[kk & 1];
            ABFrag bfr;
            load_b(bfr, wcur, (wave * 16 + ln16) * WP_S, half);
#pragma unroll
            for (int mt = 0; mt < 4; ++mt) {
                ABFrag afr;
                load_a(afr, xs, (mt * 16 + ln16) * XS_S, kk * 32, half);
                wmma_bf16(acc1[mt], afr, bfr);
            }
        }

        // ---- epilogue: +b1, silu, scale by routing weight, bf16 -> hb ----
        {
            float b1v = b1reg[c];
#pragma unroll
            for (int mt = 0; mt < 4; ++mt) {
                int m0 = mt * 16 + 8 * half;
                float4 wv0 = *(const float4*)&wrow[m0];
                float4 wv1 = *(const float4*)&wrow[m0 + 4];
                float wv[8] = { wv0.x, wv0.y, wv0.z, wv0.w,
                                wv1.x, wv1.y, wv1.z, wv1.w };
#pragma unroll
                for (int r = 0; r < 8; ++r) {
                    float v = fast_silu(acc1[mt].f[r] + b1v) * wv[r];
                    hb[(m0 + r) * HB_S + wave * 16 + ln16] = f2bf(v);
                }
            }
        }

        // ---------------- GEMM2 over K = FC ----------------
        for (int j = 0; j < FC / 32; ++j) {
            __syncthreads();              // hb written / previous w2p done
            {   // stage W2 panel [512 n][32 k], transposed on LDS write
#pragma unroll
                for (int g = 0; g < 4; ++g) {
                    uint2 d[4];
#pragma unroll
                    for (int i = 0; i < 4; ++i) {
                        int k = w2kr + 2 * (g * 4 + i);
                        const unsigned short* src =
                            w2bf + ((size_t)e * FF + (size_t)(ff0 + j * 32 + k)) * H
                                 + w2nblk * 4;
                        d[i] = *(const uint2*)src;
                    }
#pragma unroll
                    for (int i = 0; i < 4; ++i) {
                        int k = w2kr + 2 * (g * 4 + i);
                        const unsigned short* ds = (const unsigned short*)&d[i];
#pragma unroll
                        for (int q = 0; q < 4; ++q)
                            w2p[(w2nblk * 4 + q) * WP_S + k] = ds[q];
                    }
                }
            }
            __syncthreads();
            ABFrag afr[4];
#pragma unroll
            for (int mt = 0; mt < 4; ++mt)
                load_a(afr[mt], hb, (mt * 16 + ln16) * HB_S, j * 32, half);
#pragma unroll
            for (int nt = 0; nt < 4; ++nt) {
                ABFrag bfr;
                load_b(bfr, w2p, (wave * 64 + nt * 16 + ln16) * WP_S, half);
#pragma unroll
                for (int mt = 0; mt < 4; ++mt)
                    wmma_bf16(acc2[mt][nt], afr[mt], bfr);
            }
        }
    }

    // ---- scatter-accumulate into out (rows unique within this launch) ----
#pragma unroll
    for (int mt = 0; mt < 4; ++mt)
#pragma unroll
        for (int r = 0; r < 8; ++r) {
            int m = mt * 16 + r + 8 * half;
            int tok = trow[m];
            if (tok >= 0) {
                float* p = out + (size_t)tok * H + wave * 64;
#pragma unroll
                for (int nt = 0; nt < 4; ++nt)
                    p[nt * 16 + ln16] += acc2[mt][nt].f[r];
            }
        }
}

// ---------------------------------------------------------------------------
extern "C" void kernel_launch(void* const* d_in, const int* in_sizes, int n_in,
                              void* d_out, int out_size, void* d_ws, size_t ws_size,
                              hipStream_t stream)
{
    const float* x  = (const float*)d_in[0];
    const float* Wg = (const float*)d_in[1];
    const float* bg = (const float*)d_in[2];
    const float* W1 = (const float*)d_in[3];
    const float* b1 = (const float*)d_in[4];
    const float* W2 = (const float*)d_in[5];
    const float* b2 = (const float*)d_in[6];
    float* out = (float*)d_out;

    char* ws = (char*)d_ws;
    size_t off = 0;
    auto carve = [&](size_t bytes) -> void* {
        void* p = ws + off;
        off += (bytes + 255) & ~(size_t)255;
        return p;
    };
    unsigned short* xbf  = (unsigned short*)carve((size_t)NTOK * H * 2);      // 16 MB
    unsigned short* w1bf = (unsigned short*)carve((size_t)NE * H * FF * 2);   // 16 MB
    unsigned short* w2bf = (unsigned short*)carve((size_t)NE * FF * H * 2);   // 16 MB
    int*   tok_list = (int*)  carve((size_t)NE * NTOK * 4);
    float* w_list   = (float*)carve((size_t)NE * NTOK * 4);
    int*   cnt      = (int*)  carve(64);

    k_zero<<<1, 32, 0, stream>>>(cnt, NE);

    int n4 = NE * H * FF / 4;   // W1 and W2 have identical element counts
    k_f32_to_bf16<<<(n4 + 255) / 256, 256, 0, stream>>>(W1, w1bf, n4);
    k_f32_to_bf16<<<(n4 + 255) / 256, 256, 0, stream>>>(W2, w2bf, n4);

    k_gate<<<NTOK / 8, 256, 0, stream>>>(x, Wg, bg, b2, xbf,
                                         tok_list, w_list, cnt, out);

    for (int e = 0; e < NE; ++e)
        k_expert<<<NTOK / MT, 256, 0, stream>>>(xbf, w1bf, w2bf, b1,
                                                tok_list, w_list, cnt, out, e);
}
